// MPNNet_v2_29635274342995
// MI455X (gfx1250) — compile-verified, matrix operand order checked by reading, not run on hardware
//
#include <hip/hip_runtime.h>
#include <hip/hip_bf16.h>
#include <math.h>

// ---------------- problem constants (from reference) ----------------
#define N_NODES 30000
#define N_EDGES 60000
#define NB      512
#define DIM     64
#define FEAT    14
#define EDIM    4
#define NSTEM   8
#define NOPS    105
#define EWC     4096                 // DIM*DIM
#define NSEL    (NB * NSTEM)        // 4096 gathered head rows

typedef __attribute__((ext_vector_type(16))) _Float16 v16h;
typedef __attribute__((ext_vector_type(8)))  _Float16 f16x8;
typedef __attribute__((ext_vector_type(2)))  _Float16 f16x2;
typedef __attribute__((ext_vector_type(8)))  float    v8f;

__device__ __forceinline__ float lrelu(float v){ return v > 0.0f ? v : 0.01f * v; }
__device__ __forceinline__ float sigm(float v){ return 1.0f / (1.0f + expf(-v)); }

// WMMA 16x16x32 f16 fragment slice for this lane: per ISA 7.12.2 the 16 halves a
// lane holds are two contiguous 8-half runs (K+0..7 and K+16..23, lanes>=16 get +8).
__device__ __forceinline__ v16h ldfrag(const _Float16* p){
  f16x8 lo = *(const f16x8*)p;
  f16x8 hi = *(const f16x8*)(p + 16);
  v16h r;
  #pragma unroll
  for (int i = 0; i < 8; ++i){ r[i] = lo[i]; r[8 + i] = hi[i]; }
  return r;
}

// order-preserving float <-> uint for atomicMax-based segment max
__device__ __forceinline__ unsigned enc_ord(float f){
  unsigned u = __float_as_uint(f);
  return (u & 0x80000000u) ? ~u : (u | 0x80000000u);
}
__device__ __forceinline__ float dec_ord(unsigned u){
  unsigned b = (u & 0x80000000u) ? (u & 0x7fffffffu) : ~u;
  return __uint_as_float(b);
}

// ---------------- weight conversion kernels ----------------
__global__ void k_f2h(const float* __restrict__ s, _Float16* __restrict__ d, int n){
  int i = blockIdx.x * 256 + threadIdx.x;
  if (i < n) d[i] = (_Float16)s[i];
}
// lin2_w [105,512] -> f16 padded to [112,512]
__global__ void k_pad_lin2(const float* __restrict__ s, _Float16* __restrict__ d){
  int i = blockIdx.x * 256 + threadIdx.x;        // 112*512
  if (i >= 112 * 512) return;
  int o = i >> 9, k = i & 511;
  d[i] = (o < NOPS) ? (_Float16)s[o * 512 + k] : (_Float16)0.0f;
}

// ---------------- lin0 + state init ----------------
__global__ void k_lin0(const float* __restrict__ x, const float* __restrict__ w,
                       const float* __restrict__ b, float* __restrict__ out,
                       float* __restrict__ h, _Float16* __restrict__ out_h){
  int idx = blockIdx.x * 256 + threadIdx.x;      // N*64 exact
  int n = idx >> 6, f = idx & 63;
  float acc = b[f];
  #pragma unroll
  for (int k = 0; k < FEAT; ++k) acc += x[n * FEAT + k] * w[f * FEAT + k];
  float v = lrelu(acc);
  out[idx] = v; h[idx] = v; out_h[idx] = (_Float16)v;
}

// ---------------- edge hidden: eh[E,128] = lrelu(edge_attr @ e1_w.T + b) (f16) ----------------
__global__ void k_eh(const float* __restrict__ ea, const float* __restrict__ w,
                     const float* __restrict__ b, _Float16* __restrict__ eh){
  int idx = blockIdx.x * 256 + threadIdx.x;      // E*128 exact
  int e = idx >> 7, j = idx & 127;
  float acc = b[j];
  #pragma unroll
  for (int k = 0; k < EDIM; ++k) acc += ea[e * EDIM + k] * w[j * EDIM + k];
  eh[idx] = (_Float16)lrelu(acc);
}

// ---------------- big edge GEMM: ew[E,4096] = eh[E,128] @ e2_w.T + e2_b (WMMA, f16 out) ------
// grid (E/16, 16), block 128 (4 waves). wave -> 64-column strip, 4 accumulators, K=128.
__global__ void k_edge_gemm(const _Float16* __restrict__ eh, const _Float16* __restrict__ w,
                            const float* __restrict__ bias, _Float16* __restrict__ ew){
  int wave = threadIdx.x >> 5, lane = threadIdx.x & 31;
  int mtile = blockIdx.x;
  int n0 = (blockIdx.y * 4 + wave) * 64;
  int l15 = lane & 15;
  int khalf = (lane >= 16) ? 8 : 0;
  const _Float16* arow = eh + (size_t)(mtile * 16 + l15) * 128;
  v8f zero = {};
  v8f acc[4]; acc[0] = zero; acc[1] = zero; acc[2] = zero; acc[3] = zero;
  #pragma unroll
  for (int k0 = 0; k0 < 128; k0 += 32) {
    v16h a = ldfrag(arow + k0 + khalf);
    #pragma unroll
    for (int t = 0; t < 4; ++t) {
      int n = n0 + t * 16 + l15;
      v16h bf = ldfrag(w + (size_t)n * 128 + k0 + khalf);
      acc[t] = __builtin_amdgcn_wmma_f32_16x16x32_f16(false, a, false, bf,
                                                      (short)0, acc[t], false, false);
    }
  }
  int mbase = mtile * 16 + ((lane >= 16) ? 8 : 0);
  #pragma unroll
  for (int t = 0; t < 4; ++t) {
    int n = n0 + t * 16 + l15;
    float bv = bias[n];
    #pragma unroll
    for (int r = 0; r < 8; ++r)
      ew[(size_t)(mbase + r) * EWC + n] = (_Float16)(acc[t][r] + bv);
  }
}

// ---------------- degree ----------------
__global__ void k_deg(const int* __restrict__ ei, int* __restrict__ degc){
  int e = blockIdx.x * 256 + threadIdx.x;
  if (e < N_EDGES) atomicAdd(&degc[ei[N_EDGES + e]], 1);
}
__global__ void k_invdeg(const int* __restrict__ degc, float* __restrict__ invdeg){
  int n = blockIdx.x * 256 + threadIdx.x;
  if (n < N_NODES) invdeg[n] = 1.0f / (float)max(degc[n], 1);
}

// ---------------- per-edge matvec + scatter add (memory-bound sweep over ew) ----------------
// block = 256 threads = 8 waves = 8 edges; E/8 = 7500 blocks exact.
__global__ void k_msg(const float* __restrict__ out, const _Float16* __restrict__ ew,
                      const int* __restrict__ ei, float* __restrict__ agg){
  __shared__ float s_o[8][DIM];
  int wave = threadIdx.x >> 5, lane = threadIdx.x & 31;
  int e = blockIdx.x * 8 + wave;
  int src = ei[e], dst = ei[N_EDGES + e];
  s_o[wave][lane]      = out[(size_t)src * DIM + lane];
  s_o[wave][lane + 32] = out[(size_t)src * DIM + lane + 32];
  __syncthreads();
  const f16x2* wp = (const f16x2*)(ew + (size_t)e * EWC);
  float a0 = 0.0f, a1 = 0.0f;
  #pragma unroll 8
  for (int d = 0; d < DIM; ++d) {
    float ov = s_o[wave][d];
    f16x2 w2 = wp[d * 32 + lane];           // features 2*lane, 2*lane+1 -- coalesced
    a0 += ov * (float)w2.x;
    a1 += ov * (float)w2.y;
  }
  atomicAdd(&agg[(size_t)dst * DIM + 2 * lane],     a0);
  atomicAdd(&agg[(size_t)dst * DIM + 2 * lane + 1], a1);
}

// ---------------- fused NNConv-root + mean-aggr + GRU (block per node, 64 threads) ----------
__global__ void k_node(const float* __restrict__ agg, const float* __restrict__ invdeg,
                       const float* __restrict__ cr, const float* __restrict__ cb,
                       const float* __restrict__ wih, const float* __restrict__ whh,
                       const float* __restrict__ bih, const float* __restrict__ bhh,
                       float* __restrict__ out, float* __restrict__ h,
                       _Float16* __restrict__ out_h){
  __shared__ float s_o[DIM], s_h[DIM], s_m[DIM];
  int n = blockIdx.x, t = threadIdx.x;
  size_t base = (size_t)n * DIM;
  float hv = h[base + t];
  s_o[t] = out[base + t];
  s_h[t] = hv;
  __syncthreads();
  float acc = cb[t] + agg[base + t] * invdeg[n];
  #pragma unroll 8
  for (int d = 0; d < DIM; ++d) acc += s_o[d] * cr[d * DIM + t];
  s_m[t] = lrelu(acc);
  __syncthreads();
  float ir = bih[t], iz = bih[64 + t], in_ = bih[128 + t];
  float hr = bhh[t], hz = bhh[64 + t], hn  = bhh[128 + t];
  #pragma unroll 4
  for (int d = 0; d < DIM; ++d) {
    float mv = s_m[d], hd = s_h[d];
    ir  += mv * wih[t * DIM + d];
    iz  += mv * wih[(64 + t) * DIM + d];
    in_ += mv * wih[(128 + t) * DIM + d];
    hr  += hd * whh[t * DIM + d];
    hz  += hd * whh[(64 + t) * DIM + d];
    hn  += hd * whh[(128 + t) * DIM + d];
  }
  float r = sigm(ir + hr), z = sigm(iz + hz);
  float nn = tanhf(in_ + r * hn);
  float hnew = (1.0f - z) * nn + z * hv;
  out[base + t] = hnew; h[base + t] = hnew; out_h[base + t] = (_Float16)hnew;
}

// ---------------- head GEMM 1 (WMMA): gather stem rows, [4096,64] @ lin1_w.T -> lrelu f16 ----
__global__ void k_head1(const _Float16* __restrict__ out_h, const int* __restrict__ stem,
                        const _Float16* __restrict__ w, const float* __restrict__ bias,
                        _Float16* __restrict__ hbuf){
  int wave = threadIdx.x >> 5, lane = threadIdx.x & 31;
  int mtile = blockIdx.x;
  int ntile = blockIdx.y * 4 + wave;
  int l15 = lane & 15;
  int khalf = (lane >= 16) ? 8 : 0;
  int node = stem[mtile * 16 + l15];
  int ncol = ntile * 16 + l15;
  v8f acc = {};
  #pragma unroll
  for (int k0 = 0; k0 < DIM; k0 += 32) {
    v16h a = ldfrag(out_h + (size_t)node * DIM + k0 + khalf);
    v16h b = ldfrag(w + (size_t)ncol * DIM + k0 + khalf);
    acc = __builtin_amdgcn_wmma_f32_16x16x32_f16(false, a, false, b, (short)0, acc, false, false);
  }
  float bv = bias[ncol];
  int mbase = mtile * 16 + ((lane >= 16) ? 8 : 0);
  #pragma unroll
  for (int r = 0; r < 8; ++r)
    hbuf[(size_t)(mbase + r) * 512 + ncol] = (_Float16)lrelu(acc[r] + bv);
}

// ---------------- head GEMM 2 (WMMA): [4096,512] @ lin2_w.T + b -> add_logits in d_out ------
__global__ void k_head2(const _Float16* __restrict__ hbuf, const _Float16* __restrict__ w,
                        const float* __restrict__ bias, float* __restrict__ logits){
  int lane = threadIdx.x & 31;
  int mtile = blockIdx.x, ntile = blockIdx.y;    // 256 x 7
  int l15 = lane & 15;
  int khalf = (lane >= 16) ? 8 : 0;
  int arow = mtile * 16 + l15;
  int ncol = ntile * 16 + l15;                   // 0..111 (padded weights)
  v8f acc = {};
  #pragma unroll
  for (int k0 = 0; k0 < 512; k0 += 32) {
    v16h a = ldfrag(hbuf + (size_t)arow * 512 + k0 + khalf);
    v16h b = ldfrag(w + (size_t)ncol * 512 + k0 + khalf);
    acc = __builtin_amdgcn_wmma_f32_16x16x32_f16(false, a, false, b, (short)0, acc, false, false);
  }
  if (ncol < NOPS) {
    float bv = bias[ncol];
    int mbase = mtile * 16 + ((lane >= 16) ? 8 : 0);
    #pragma unroll
    for (int r = 0; r < 8; ++r)
      logits[(size_t)(mbase + r) * NOPS + ncol] = acc[r] + bv;
  }
}

// ---------------- Set2Set ----------------
__global__ void k_lstm(const float* __restrict__ qstar, const float* __restrict__ wih,
                       const float* __restrict__ whh, const float* __restrict__ bih,
                       const float* __restrict__ bhh, float* __restrict__ hh,
                       float* __restrict__ cc){
  __shared__ float s_q[2 * DIM], s_h[DIM], s_g[4 * DIM];
  int b = blockIdx.x, t = threadIdx.x;           // 256 threads
  if (t < 128) s_q[t] = qstar[b * 128 + t];
  if (t < 64)  s_h[t] = hh[b * 64 + t];
  __syncthreads();
  float g = bih[t] + bhh[t];
  #pragma unroll 4
  for (int k = 0; k < 128; ++k) g += s_q[k] * wih[t * 128 + k];
  #pragma unroll 4
  for (int k = 0; k < 64; ++k)  g += s_h[k] * whh[t * 64 + k];
  s_g[t] = g;
  __syncthreads();
  if (t < 64) {
    float ii = s_g[t], ff = s_g[64 + t], gg = s_g[128 + t], oo = s_g[192 + t];
    float c = sigm(ff) * cc[b * 64 + t] + sigm(ii) * tanhf(gg);
    cc[b * 64 + t] = c;
    hh[b * 64 + t] = sigm(oo) * tanhf(c);
  }
}

__global__ void k_att_e(const float* __restrict__ out, const float* __restrict__ hh,
                        const int* __restrict__ batch, float* __restrict__ e_ws,
                        unsigned* __restrict__ emaxe){
  int n = blockIdx.x * 256 + threadIdx.x;
  if (n >= N_NODES) return;
  int b = batch[n];
  float e = 0.0f;
  #pragma unroll 8
  for (int k = 0; k < DIM; ++k) e += out[(size_t)n * DIM + k] * hh[b * DIM + k];
  e_ws[n] = e;
  atomicMax(&emaxe[b], enc_ord(e));
}
__global__ void k_att_ex(const float* __restrict__ e_ws, const unsigned* __restrict__ emaxe,
                         const int* __restrict__ batch, float* __restrict__ ex_ws,
                         float* __restrict__ den){
  int n = blockIdx.x * 256 + threadIdx.x;
  if (n >= N_NODES) return;
  int b = batch[n];
  float ex = expf(e_ws[n] - dec_ord(emaxe[b]));
  ex_ws[n] = ex;
  atomicAdd(&den[b], ex);
}
__global__ void k_att_r(const float* __restrict__ out, const float* __restrict__ ex_ws,
                        const float* __restrict__ den, const int* __restrict__ batch,
                        float* __restrict__ rvec){
  int idx = blockIdx.x * 256 + threadIdx.x;      // N*64 exact
  int n = idx >> 6, f = idx & 63;
  int b = batch[n];
  float a = ex_ws[n] / den[b];
  atomicAdd(&rvec[b * DIM + f], a * out[idx]);
}
__global__ void k_qstar(const float* __restrict__ hh, const float* __restrict__ rvec,
                        float* __restrict__ qstar){
  int idx = blockIdx.x * 256 + threadIdx.x;      // B*128 exact
  int b = idx >> 7, f = idx & 127;
  qstar[idx] = (f < 64) ? hh[b * 64 + f] : rvec[b * 64 + (f - 64)];
}
__global__ void k_lin3(const float* __restrict__ qstar, const float* __restrict__ w,
                       const float* __restrict__ b, float* __restrict__ r_out){
  int bi = blockIdx.x * 64 + threadIdx.x;
  if (bi >= NB) return;
  float acc = b[0];
  #pragma unroll 4
  for (int k = 0; k < 128; ++k) acc += qstar[bi * 128 + k] * w[k];
  r_out[bi] = acc;
}

// ---------------- host launch ----------------
extern "C" void kernel_launch(void* const* d_in, const int* in_sizes, int n_in,
                              void* d_out, int out_size, void* d_ws, size_t ws_size,
                              hipStream_t stream) {
  (void)in_sizes; (void)n_in; (void)out_size; (void)ws_size;
  const float* x         = (const float*)d_in[0];
  const float* edge_attr = (const float*)d_in[1];
  const int*   ei        = (const int*)d_in[2];
  const int*   batch     = (const int*)d_in[3];
  const int*   stem      = (const int*)d_in[4];
  const float* lin0_w    = (const float*)d_in[5];
  const float* lin0_b    = (const float*)d_in[6];
  const float* e1_w      = (const float*)d_in[7];
  const float* e1_b      = (const float*)d_in[8];
  const float* e2_w      = (const float*)d_in[9];
  const float* e2_b      = (const float*)d_in[10];
  const float* conv_root = (const float*)d_in[11];
  const float* conv_b    = (const float*)d_in[12];
  const float* gwih      = (const float*)d_in[13];
  const float* gwhh      = (const float*)d_in[14];
  const float* gbih      = (const float*)d_in[15];
  const float* gbhh      = (const float*)d_in[16];
  const float* l1w       = (const float*)d_in[17];
  const float* l1b       = (const float*)d_in[18];
  const float* l2w       = (const float*)d_in[19];
  const float* l2b       = (const float*)d_in[20];
  const float* lwih      = (const float*)d_in[21];
  const float* lwhh      = (const float*)d_in[22];
  const float* lbih      = (const float*)d_in[23];
  const float* lbhh      = (const float*)d_in[24];
  const float* l3w       = (const float*)d_in[25];
  const float* l3b       = (const float*)d_in[26];
  float* out = (float*)d_out;                       // [0,512): r_out ; [512,...): add_logits

  char* ws = (char*)d_ws;
  size_t off = 0;
  auto take = [&](size_t bytes) -> char* {
    off = (off + 255) & ~(size_t)255;
    char* p = ws + off; off += bytes; return p;
  };
  _Float16* ew_h   = (_Float16*)take((size_t)N_EDGES * EWC * 2);   // 469 MB, the sweep buffer
  _Float16* eh     = (_Float16*)take((size_t)N_EDGES * 128 * 2);
  _Float16* e2wh   = (_Float16*)take((size_t)EWC * 128 * 2);
  _Float16* l1wh   = (_Float16*)take((size_t)512 * 64 * 2);
  _Float16* l2wh   = (_Float16*)take((size_t)112 * 512 * 2);
  float*    nout   = (float*)take((size_t)N_NODES * DIM * 4);
  float*    nh     = (float*)take((size_t)N_NODES * DIM * 4);
  _Float16* nout_h = (_Float16*)take((size_t)N_NODES * DIM * 2);
  float*    agg    = (float*)take((size_t)N_NODES * DIM * 4);
  int*      degc   = (int*)take((size_t)N_NODES * 4);
  float*    invdeg = (float*)take((size_t)N_NODES * 4);
  _Float16* hbuf   = (_Float16*)take((size_t)NSEL * 512 * 2);
  float*    e_ws   = (float*)take((size_t)N_NODES * 4);
  float*    ex_ws  = (float*)take((size_t)N_NODES * 4);
  unsigned* emaxe  = (unsigned*)take((size_t)NB * 4);
  float*    den    = (float*)take((size_t)NB * 4);
  float*    rvec   = (float*)take((size_t)NB * DIM * 4);
  float*    qstar  = (float*)take((size_t)NB * 2 * DIM * 4);
  float*    hh     = (float*)take((size_t)NB * DIM * 4);
  float*    cc     = (float*)take((size_t)NB * DIM * 4);

  // weight conversions (f32 -> f16)
  k_f2h<<<(EWC * 128 + 255) / 256, 256, 0, stream>>>(e2_w, e2wh, EWC * 128);
  k_f2h<<<(512 * 64 + 255) / 256, 256, 0, stream>>>(l1w, l1wh, 512 * 64);
  k_pad_lin2<<<(112 * 512 + 255) / 256, 256, 0, stream>>>(l2w, l2wh);

  // node / edge feature encoders
  k_lin0<<<(N_NODES * DIM) / 256, 256, 0, stream>>>(x, lin0_w, lin0_b, nout, nh, nout_h);
  k_eh<<<(N_EDGES * 128) / 256, 256, 0, stream>>>(edge_attr, e1_w, e1_b, eh);

  // big edge-conditioned weight GEMM (WMMA f16)
  k_edge_gemm<<<dim3(N_EDGES / 16, 16), 128, 0, stream>>>(eh, e2wh, e2_b, ew_h);

  // degrees
  hipMemsetAsync(degc, 0, (size_t)N_NODES * 4, stream);
  k_deg<<<(N_EDGES + 255) / 256, 256, 0, stream>>>(ei, degc);
  k_invdeg<<<(N_NODES + 255) / 256, 256, 0, stream>>>(degc, invdeg);

  // 12 message-passing iterations (memory-bound on ew_h sweep)
  for (int it = 0; it < 12; ++it) {
    hipMemsetAsync(agg, 0, (size_t)N_NODES * DIM * 4, stream);
    k_msg<<<N_EDGES / 8, 256, 0, stream>>>(nout, ew_h, ei, agg);
    k_node<<<N_NODES, 64, 0, stream>>>(agg, invdeg, conv_root, conv_b,
                                       gwih, gwhh, gbih, gbhh, nout, nh, nout_h);
  }

  // per-stem head (WMMA on 4096 gathered rows only)
  k_head1<<<dim3(NSEL / 16, 8), 128, 0, stream>>>(nout_h, stem, l1wh, l1b, hbuf);
  k_head2<<<dim3(NSEL / 16, 7), 32, 0, stream>>>(hbuf, l2wh, l2b, out + NB);

  // Set2Set (3 steps)
  hipMemsetAsync(qstar, 0, (size_t)NB * 128 * 4, stream);
  hipMemsetAsync(hh, 0, (size_t)NB * DIM * 4, stream);
  hipMemsetAsync(cc, 0, (size_t)NB * DIM * 4, stream);
  for (int s = 0; s < 3; ++s) {
    hipMemsetAsync(emaxe, 0, (size_t)NB * 4, stream);
    hipMemsetAsync(den, 0, (size_t)NB * 4, stream);
    hipMemsetAsync(rvec, 0, (size_t)NB * DIM * 4, stream);
    k_lstm<<<NB, 256, 0, stream>>>(qstar, lwih, lwhh, lbih, lbhh, hh, cc);
    k_att_e<<<(N_NODES + 255) / 256, 256, 0, stream>>>(nout, hh, batch, e_ws, emaxe);
    k_att_ex<<<(N_NODES + 255) / 256, 256, 0, stream>>>(e_ws, emaxe, batch, ex_ws, den);
    k_att_r<<<(N_NODES * DIM) / 256, 256, 0, stream>>>(nout, ex_ws, den, batch, rvec);
    k_qstar<<<(NB * 128) / 256, 256, 0, stream>>>(hh, rvec, qstar);
  }
  k_lin3<<<(NB + 63) / 64, 64, 0, stream>>>(qstar, l3w, l3b, out);
}
